// CustomGATModel_EdgeAware_77094662963241
// MI455X (gfx1250) — matrix operand order repaired and checked by reading, not compiled, not run on hardware
//
#include <hip/hip_runtime.h>

#define DEV_INLINE __device__ __forceinline__

// Problem constants (match reference)
constexpr int B_ = 128, W_ = 128, FIN_ = 64, N_ = 64, E_ = 4096, ED_ = 16;
constexpr int H_ = 4, C_ = 32, HC_ = 128, FLAT_ = 8192, HID_ = 2048;

typedef __attribute__((ext_vector_type(16))) __bf16 v16bf;
typedef __attribute__((ext_vector_type(8)))  __bf16 v8bf;
typedef __attribute__((ext_vector_type(8)))  float  v8f;

// ---------- helpers ----------
// native converts on gfx1250 (v_cvt_*bf16*), RNE — replaces manual bit fiddling
DEV_INLINE __bf16 f32_to_bf16(float f) { return (__bf16)f; }

// order-preserving float<->uint key (for atomicMax-based segment max)
DEV_INLINE unsigned enc_f32(float f) {
  unsigned u = __float_as_uint(f);
  return (u & 0x80000000u) ? ~u : (u | 0x80000000u);
}
DEV_INLINE float dec_f32(unsigned k) {
  unsigned u = (k & 0x80000000u) ? (k ^ 0x80000000u) : ~k;
  return __uint_as_float(u);
}

// A-matrix 16-bit 16x32 layout (ISA 7.12.2): lane holds row m=lane%16,
// K = {8*half+0..7} in v[0..3] halves and {16+8*half+0..7} in v[4..7].
DEV_INLINE v16bf load_a16(const __bf16* p, int half) {
  p += half * 8;
  v8bf lo = *(const v8bf*)p;
  v8bf hi = *(const v8bf*)(p + 16);
  return __builtin_shufflevector(lo, hi, 0,1,2,3,4,5,6,7,8,9,10,11,12,13,14,15);
}
// B-matrix 16-bit 32x16 layout: lane holds column n=lane%16,
// K = {8*half+0..7} then {16+8*half+0..7} (two b128 loads, matching A).
DEV_INLINE v16bf load_b16(const __bf16* p, int half) {
  p += half * 8;
  v8bf lo = *(const v8bf*)p;
  v8bf hi = *(const v8bf*)(p + 16);
  return __builtin_shufflevector(lo, hi, 0,1,2,3,4,5,6,7,8,9,10,11,12,13,14,15);
}
// A tile from an f32 source, converting in-register
DEV_INLINE v16bf load_a16_f32(const float* p, int half) {
  p += half * 8;
  v16bf r;
#pragma unroll
  for (int i = 0; i < 8; ++i) {
    r[i]     = f32_to_bf16(p[i]);
    r[8 + i] = f32_to_bf16(p[16 + i]);
  }
  return r;
}

// ---------- prep kernels ----------
__global__ void cvt_bf16_kernel(const float* __restrict__ s, __bf16* __restrict__ d, int n) {
  for (int i = blockIdx.x * blockDim.x + threadIdx.x; i < n; i += gridDim.x * blockDim.x)
    d[i] = f32_to_bf16(s[i]);
}

__global__ void init_kernel(unsigned* __restrict__ mkey, float* __restrict__ den,
                            float* __restrict__ gat) {
  int i = blockIdx.x * blockDim.x + threadIdx.x;            // grid covers B*N*HC
  if (i < B_ * N_ * H_) { mkey[i] = enc_f32(-__builtin_huge_valf()); den[i] = 0.f; }
  if (i < B_ * N_ * HC_) gat[i] = 0.f;
}

// ---------- TCN: Conv1d(F_IN->N, k=3, pad=1) over window dim ----------
__global__ void tcn_kernel(const float* __restrict__ x, const float* __restrict__ tw,
                           const float* __restrict__ tb, float* __restrict__ xg) {
  __shared__ float xs[W_][FIN_];                            // 32 KB of the 320 KB LDS
  int b = blockIdx.x;
  for (int i = threadIdx.x; i < W_ * FIN_; i += blockDim.x)
    xs[i / FIN_][i % FIN_] = x[(size_t)b * W_ * FIN_ + i];
  __syncthreads();
  int w  = threadIdx.x & (W_ - 1);
  int nh = threadIdx.x >> 7;                                // 0..1
  for (int n = nh; n < N_; n += 2) {
    float acc = tb[n];
    const float* wp = tw + (size_t)n * FIN_ * 3;
#pragma unroll 4
    for (int f = 0; f < FIN_; ++f) {
      float w0 = wp[f * 3 + 0], w1 = wp[f * 3 + 1], w2 = wp[f * 3 + 2];
      float xm = (w > 0)      ? xs[w - 1][f] : 0.f;
      float xc = xs[w][f];
      float xp = (w < W_ - 1) ? xs[w + 1][f] : 0.f;
      acc += xm * w0 + xc * w1 + xp * w2;
    }
    xg[((size_t)b * N_ + n) * W_ + w] = acc;
  }
}

// ---------- lin_l: per-batch [64x128] @ [128x128] via WMMA bf16 ----------
__global__ void lin_l_kernel(const float* __restrict__ xg, const __bf16* __restrict__ wl,
                             const float* __restrict__ lb, float* __restrict__ h) {
  int b = blockIdx.x;
  int wave = threadIdx.x >> 5, lane = threadIdx.x & 31;
  int half = lane >> 4, l16 = lane & 15;
  for (int t = wave; t < 32; t += 8) {                      // 4x8 tiles, 8 waves
    int m0 = (t >> 3) * 16, n0 = (t & 7) * 16;
    const float*  ap = xg + ((size_t)b * N_ + m0 + l16) * W_;
    const __bf16* bp = wl + (size_t)(n0 + l16) * W_;
    v8f acc = {};
#pragma unroll
    for (int k0 = 0; k0 < W_; k0 += 32) {
      v16bf a = load_a16_f32(ap + k0, half);
      v16bf bb = load_b16(bp + k0, half);
      acc = __builtin_amdgcn_wmma_f32_16x16x32_bf16(false, a, false, bb,
                                                    (short)0, acc, false, false);
    }
    float bias = lb[n0 + l16];
#pragma unroll
    for (int r = 0; r < 8; ++r) {
      int m = m0 + r + 8 * half;
      h[((size_t)b * N_ + m) * HC_ + n0 + l16] = acc[r] + bias;
    }
  }
}

// ---------- edge pass 1: score + segment max (wave per edge) ----------
__global__ void edge_score_kernel(const int* __restrict__ ei, const float* __restrict__ ea,
                                  const float* __restrict__ lew, const float* __restrict__ leb,
                                  const float* __restrict__ att, const float* __restrict__ h,
                                  float* __restrict__ sc, unsigned* __restrict__ mkey) {
  int gw = blockIdx.x * 8 + (threadIdx.x >> 5);
  int lane = threadIdx.x & 31;
  int b = gw >> 12, e = gw & (E_ - 1);
  int src = ei[e], dst = ei[E_ + e];
  const float* eap = ea + ((size_t)b * E_ + e) * ED_;
  float eav[ED_];
#pragma unroll
  for (int d = 0; d < ED_; ++d) eav[d] = eap[d];
#pragma unroll
  for (int j = 0; j < H_; ++j) {
    int hc = j * C_ + lane;
    float ef = leb[hc];
    const float* wp = lew + (size_t)hc * ED_;
#pragma unroll
    for (int d = 0; d < ED_; ++d) ef += eav[d] * wp[d];
    float xi = h[((size_t)b * N_ + dst) * HC_ + hc];
    float xj = h[((size_t)b * N_ + src) * HC_ + hc];
    float a = xi + xj + ef;
    a = (a > 0.f) ? a : 0.01f * a;                          // leaky_relu
    float p = a * att[j * C_ + lane];
#pragma unroll
    for (int off = 16; off; off >>= 1) p += __shfl_xor(p, off, 32);
    if (lane == 0) {
      sc[((size_t)b * E_ + e) * H_ + j] = p;
      atomicMax(&mkey[((size_t)b * N_ + dst) * H_ + j], enc_f32(p));
    }
  }
}

// ---------- edge pass 2: exp + segment sum ----------
__global__ void edge_exp_kernel(const int* __restrict__ ei, float* __restrict__ sc,
                                const unsigned* __restrict__ mkey, float* __restrict__ den) {
  size_t i = (size_t)blockIdx.x * blockDim.x + threadIdx.x; // B*E*H threads
  int j = i & (H_ - 1);
  size_t be = i >> 2;
  int e = be & (E_ - 1), b = be >> 12;
  int dst = ei[E_ + e];
  float mf = dec_f32(mkey[((size_t)b * N_ + dst) * H_ + j]);
  float ex = __expf(sc[i] - mf);
  sc[i] = ex;
  atomicAdd(&den[((size_t)b * N_ + dst) * H_ + j], ex);
}

// ---------- edge pass 3: alpha-weighted scatter (wave per edge) ----------
__global__ void edge_scatter_kernel(const int* __restrict__ ei, const float* __restrict__ sc,
                                    const float* __restrict__ den, const float* __restrict__ h,
                                    float* __restrict__ gat) {
  int gw = blockIdx.x * 8 + (threadIdx.x >> 5);
  int lane = threadIdx.x & 31;
  int b = gw >> 12, e = gw & (E_ - 1);
  int src = ei[e], dst = ei[E_ + e];
#pragma unroll
  for (int j = 0; j < H_; ++j) {
    float alpha = sc[((size_t)b * E_ + e) * H_ + j] /
                  (den[((size_t)b * N_ + dst) * H_ + j] + 1e-16f);
    int hc = j * C_ + lane;
    float msg = h[((size_t)b * N_ + src) * HC_ + hc] * alpha;
    atomicAdd(&gat[((size_t)b * N_ + dst) * HC_ + hc], msg);
  }
}

// ---------- ELU + pack x_flat to bf16 ----------
__global__ void elu_cvt_kernel(const float* __restrict__ gat, __bf16* __restrict__ abf) {
  int i = blockIdx.x * blockDim.x + threadIdx.x;            // B*FLAT threads
  float v = gat[i];
  v = (v > 0.f) ? v : (__expf(v) - 1.f);
  abf[i] = f32_to_bf16(v);
}

// ---------- fc1: [128x8192] @ [8192x2048] WMMA bf16, fused BN+ReLU ----------
// A panel (16 rows, shared by all 8 waves of the block) is staged in LDS via
// async global->LDS copies (ASYNCcnt), double-buffered against the WMMAs.
__global__ void fc1_kernel(const __bf16* __restrict__ A, const __bf16* __restrict__ Wt,
                           const float* __restrict__ bias, const float* __restrict__ bng,
                           const float* __restrict__ bnb, float* __restrict__ h1) {
  constexpr int KSTEP  = 64;   // K per stage = two WMMAs
  constexpr int APITCH = 72;   // padded pitch in halves (144B: bank stagger)
  __shared__ __bf16 As[2][16 * APITCH];

  int wave = threadIdx.x >> 5, lane = threadIdx.x & 31;
  int half = lane >> 4, l16 = lane & 31 & 15;
  int m0 = (blockIdx.x >> 4) * 16;                          // 8 M-tiles
  int n0 = ((blockIdx.x & 15) * 8 + wave) * 16;             // 128 N-tiles
  const __bf16* bp = Wt + (size_t)(n0 + (lane & 15)) * FLAT_;

  // async stagers: threads 0..127 each copy one 16B segment of the A panel
  int srow = threadIdx.x >> 3, sseg = threadIdx.x & 7;
  const __bf16* agp = A + (size_t)(m0 + srow) * FLAT_ + sseg * 8;
  unsigned lds_base = (unsigned)(size_t)&As[0][0];          // low 32b = LDS offset
  auto stage = [&](int k0, int buf) {
    if (threadIdx.x < 128) {
      unsigned laddr = lds_base + (unsigned)((buf * 16 + srow) * APITCH + sseg * 8) * 2u;
      const __bf16* g = agp + k0;
      asm volatile("global_load_async_to_lds_b128 %0, %1, off"
                   :: "v"(laddr), "v"(g) : "memory");
    }
  };

  stage(0, 0);
  asm volatile("s_wait_asynccnt 0x0" ::: "memory");
  __syncthreads();

  v8f acc0 = {}, acc1 = {};                                 // two chains for ILP
  for (int k0 = 0; k0 < FLAT_; k0 += KSTEP) {
    int buf = (k0 >> 6) & 1;
    if (k0 + KSTEP < FLAT_) stage(k0 + KSTEP, buf ^ 1);     // prefetch next panel
    __builtin_prefetch(bp + k0 + 16 * KSTEP, 0, 3);         // B stream -> global_prefetch

    const __bf16* ar = &As[buf][l16 * APITCH];
    v16bf a0 = load_a16(ar, half);                          // k-local 0..31 (ds_load)
    v16bf b0 = load_b16(bp + k0, half);
    acc0 = __builtin_amdgcn_wmma_f32_16x16x32_bf16(false, a0, false, b0,
                                                   (short)0, acc0, false, false);
    v16bf a1 = load_a16(ar + 32, half);                     // k-local 32..63
    v16bf b1 = load_b16(bp + k0 + 32, half);
    acc1 = __builtin_amdgcn_wmma_f32_16x16x32_bf16(false, a1, false, b1,
                                                   (short)0, acc1, false, false);

    if (k0 + KSTEP < FLAT_)
      asm volatile("s_wait_asynccnt 0x0" ::: "memory");     // next panel landed
    __syncthreads();                                        // all waves done with buf
  }

  int n = n0 + (lane & 15);
  float s = bng[n] * rsqrtf(1.f + 1e-5f);
  float bb = bnb[n], fb = bias[n];
#pragma unroll
  for (int r = 0; r < 8; ++r) {
    int m = m0 + r + 8 * half;
    float v = (acc0[r] + acc1[r] + fb) * s + bb;
    h1[(size_t)m * HID_ + n] = (v > 0.f) ? v : 0.f;
  }
}

// ---------- fc2: [128x2048] @ [2048x2] ----------
__global__ void fc2_kernel(const float* __restrict__ h1, const float* __restrict__ w2,
                           const float* __restrict__ b2, float* __restrict__ out) {
  int t = threadIdx.x;                                      // 256 = 128 b * 2 o
  int b = t >> 1, o = t & 1;
  float acc = b2[o];
  const float* hp = h1 + (size_t)b * HID_;
  const float* wp = w2 + (size_t)o * HID_;
  for (int k = 0; k < HID_; ++k) acc += hp[k] * wp[k];
  out[b * 2 + o] = acc;
}

// ---------- launch ----------
extern "C" void kernel_launch(void* const* d_in, const int* in_sizes, int n_in,
                              void* d_out, int out_size, void* d_ws, size_t ws_size,
                              hipStream_t stream) {
  const float* x       = (const float*)d_in[0];
  const int*   ei      = (const int*)  d_in[1];
  const float* ea      = (const float*)d_in[2];
  const float* tcn_w   = (const float*)d_in[3];
  const float* tcn_b   = (const float*)d_in[4];
  const float* lin_l_w = (const float*)d_in[5];
  const float* lin_l_b = (const float*)d_in[6];
  const float* lin_e_w = (const float*)d_in[7];
  const float* lin_e_b = (const float*)d_in[8];
  const float* att     = (const float*)d_in[9];
  const float* fc1_w   = (const float*)d_in[10];
  const float* fc1_b   = (const float*)d_in[11];
  const float* bn_g    = (const float*)d_in[12];
  const float* bn_b    = (const float*)d_in[13];
  const float* fc2_w   = (const float*)d_in[14];
  const float* fc2_b   = (const float*)d_in[15];

  // workspace carve-out (256B aligned)
  char* w = (char*)d_ws;
  size_t o = 0;
  auto take = [&](size_t bytes) -> void* {
    void* p = w + o;
    o += (bytes + 255) & ~(size_t)255;
    return p;
  };
  float*    xg   = (float*)   take(sizeof(float) * B_ * N_ * W_);    // 4 MB
  float*    h    = (float*)   take(sizeof(float) * B_ * N_ * HC_);   // 4 MB
  __bf16*   wlb  = (__bf16*)  take(sizeof(__bf16) * HC_ * W_);       // 32 KB
  float*    sc   = (float*)   take(sizeof(float) * (size_t)B_ * E_ * H_); // 8 MB
  unsigned* mkey = (unsigned*)take(sizeof(unsigned) * B_ * N_ * H_); // 128 KB
  float*    den  = (float*)   take(sizeof(float) * B_ * N_ * H_);    // 128 KB
  float*    gat  = (float*)   take(sizeof(float) * B_ * N_ * HC_);   // 4 MB
  __bf16*   abf  = (__bf16*)  take(sizeof(__bf16) * B_ * FLAT_);     // 2 MB
  __bf16*   w1b  = (__bf16*)  take(sizeof(__bf16) * (size_t)HID_ * FLAT_); // 32 MB
  float*    h1   = (float*)   take(sizeof(float) * B_ * HID_);       // 1 MB
  (void)ws_size; (void)in_sizes; (void)n_in; (void)out_size;

  dim3 blk(256);
  cvt_bf16_kernel<<<64, blk, 0, stream>>>(lin_l_w, wlb, HC_ * W_);
  cvt_bf16_kernel<<<4096, blk, 0, stream>>>(fc1_w, w1b, HID_ * FLAT_);
  init_kernel<<<(B_ * N_ * HC_) / 256, blk, 0, stream>>>(mkey, den, gat);
  tcn_kernel<<<B_, blk, 0, stream>>>(x, tcn_w, tcn_b, xg);
  lin_l_kernel<<<B_, blk, 0, stream>>>(xg, wlb, lin_l_b, h);
  edge_score_kernel<<<(B_ * E_) / 8, blk, 0, stream>>>(ei, ea, lin_e_w, lin_e_b, att, h, sc, mkey);
  edge_exp_kernel<<<(B_ * E_ * H_) / 256, blk, 0, stream>>>(ei, sc, mkey, den);
  edge_scatter_kernel<<<(B_ * E_) / 8, blk, 0, stream>>>(ei, sc, den, h, gat);
  elu_cvt_kernel<<<(B_ * FLAT_) / 256, blk, 0, stream>>>(gat, abf);
  fc1_kernel<<<128, blk, 0, stream>>>(abf, w1b, fc1_b, bn_g, bn_b, h1);
  fc2_kernel<<<1, blk, 0, stream>>>(h1, fc2_w, fc2_b, (float*)d_out);
}
